// NodeSemanticAndStructureModel_5901285065127
// MI455X (gfx1250) — matrix operand order repaired and checked by reading, not compiled, not run on hardware
//
#include <hip/hip_runtime.h>
#include <hip/hip_bf16.h>
#include <math.h>

typedef __attribute__((ext_vector_type(16))) __bf16 v16bf;
typedef __attribute__((ext_vector_type(2)))  __bf16 v2bf;
typedef __attribute__((ext_vector_type(8)))  float  v8f;

union Frag16 { uint4 u[2]; v16bf v; };

// Native f32->bf16 (RTNE) conversions; pairs fuse to v_cvt_pk_bf16_f32.
__device__ __forceinline__ unsigned pack2(float lo, float hi) {
  v2bf p;
  p.x = (__bf16)lo;
  p.y = (__bf16)hi;
  return __builtin_bit_cast(unsigned, p);
}

// ---------------------------------------------------------------------------
// Generic bf16-WMMA GEMM:  C[:, col0 + blockIdx.y*tileN ...] = epi(A*Wt^T + b)
//   A  : [M, K] f32 row-major (prolog transform applied while staging to LDS)
//   Wt : [Ncols, K] bf16 (pre-transposed weights, L2-resident)
// PRO  : 0 = raw, 1 = per-K-column affine (BN), 2 = finite-filter (+inf -> 0)
// EPI  : 0 = leaky(0.01), 1 = tanh, 2 = relu, 3 = none
// FRAGS: 16-col WMMA fragments per wave (tileN = 2*FRAGS*16; 4 -> 128, 2 -> 64)
// Block: 256 thr = 8 waves (4 M x 2 N); tile 64(M) x tileN; K step 32.
// Double-buffered LDS A tile, last K-step peeled: steady-state body is
// branch-free (no exec-masked or scalar branches around the WMMA chain).
// ---------------------------------------------------------------------------
template <int PRO, int EPI, int FRAGS>
__global__ __launch_bounds__(256) void gemm_bf16_wmma(
    const float* __restrict__ A, int M, int K,
    const __bf16* __restrict__ Wt,
    const float* __restrict__ bias,
    const float* __restrict__ kscale, const float* __restrict__ kshift,
    float* __restrict__ C, int ldc, int col0)
{
  __shared__ alignas(16) __bf16 tA[2][64 * 32];

  const int tid  = threadIdx.x;
  const int lane = tid & 31;
  const int wave = tid >> 5;
  const int wy = wave >> 1, wx = wave & 1;
  const int m0 = blockIdx.x * 64;
  const int n0 = blockIdx.y * (2 * FRAGS * 16);

  v8f acc[FRAGS] = {};

  // Staging: thread -> (row, 8-col slab). Row clamped: OOB rows compute
  // garbage that is never stored (WMMA rows are independent).
  const int arow = tid >> 2;        // 0..63
  const int acol = (tid & 3) * 8;   // 0,8,16,24
  const int grow = m0 + arow;
  const int crow = (grow < M) ? grow : (M - 1);
  const float* Arow = A + (size_t)crow * K + acol;

  auto loadTile = [&](int kk, float va[8]) {
    const float4 p0 = *(const float4*)(Arow + kk);
    const float4 p1 = *(const float4*)(Arow + kk + 4);
    va[0]=p0.x; va[1]=p0.y; va[2]=p0.z; va[3]=p0.w;
    va[4]=p1.x; va[5]=p1.y; va[6]=p1.z; va[7]=p1.w;
  };
  auto packStore = [&](int kk, float va[8], int buf) {
    #pragma unroll
    for (int j = 0; j < 8; ++j) {
      if (PRO == 1) {
        const int k = kk + acol + j;
        va[j] = kscale[k] * va[j] + kshift[k];
      } else if (PRO == 2) {
        va[j] = (fabsf(va[j]) <= 3.0e38f) ? va[j] : 0.f;  // +inf -> 0
      }
    }
    uint4 pk;
    pk.x = pack2(va[0], va[1]); pk.y = pack2(va[2], va[3]);
    pk.z = pack2(va[4], va[5]); pk.w = pack2(va[6], va[7]);
    *(uint4*)&tA[buf][arow * 32 + acol] = pk;   // one ds_store_b128
  };

  // Fragment addressing: contiguous k = 16h + s for A and B (same order in both).
  const int r   = wy * 16 + (lane & 15);
  const int kba = (lane < 16) ? 0 : 16;         // A: 16 contiguous K
  const int kbb = (lane < 16) ? 0 : 16;         // B: 16 contiguous K

  auto computeStep = [&](int s) {
    const int kk = s << 5;
    const __bf16* tb = tA[s & 1];
    Frag16 af;
    af.u[0] = *(const uint4*)&tb[r * 32 + kba];
    af.u[1] = *(const uint4*)&tb[r * 32 + kba + 8];
    Frag16 bfr[FRAGS];
    #pragma unroll
    for (int f = 0; f < FRAGS; ++f) {
      const int bc = n0 + (wx * FRAGS + f) * 16 + (lane & 15);
      const __bf16* bp = Wt + (size_t)bc * K + kk + kbb;
      bfr[f].u[0] = *(const uint4*)bp;
      bfr[f].u[1] = *(const uint4*)(bp + 8);
    }
    #pragma unroll
    for (int f = 0; f < FRAGS; ++f)
      acc[f] = __builtin_amdgcn_wmma_f32_16x16x32_bf16(
          false, af.v, false, bfr[f].v, (short)0, acc[f], false, false);
  };

  const int nsteps = K >> 5;

  // Prologue: stage tile 0.
  {
    float va[8];
    loadTile(0, va);
    packStore(0, va, 0);
  }
  __syncthreads();

  // Steady state: branch-free body, one barrier per K step.
  for (int s = 0; s < nsteps - 1; ++s) {
    const int kk = s << 5;
    float vnext[8];
    loadTile(kk + 32, vnext);                       // next tile -> regs
    __builtin_prefetch(Arow + kk + 64, 0, 3);       // speculative, never faults
    computeStep(s);                                 // WMMA chain on tile s
    packStore(kk + 32, vnext, (s + 1) & 1);         // regs -> other buffer
    __syncthreads();
  }
  // Final (peeled) step: compute only.
  computeStep(nsteps - 1);

  // ---- epilogue: bias + activation + store ----
  #pragma unroll
  for (int f = 0; f < FRAGS; ++f) {
    const int oc = n0 + (wx * FRAGS + f) * 16 + (lane & 15);
    const float b = bias[oc];
    #pragma unroll
    for (int rr = 0; rr < 8; ++rr) {
      const int row = m0 + wy * 16 + rr + ((lane & 16) ? 8 : 0);
      if (row < M) {
        float v = acc[f][rr] + b;
        if (EPI == 0)      v = (v >= 0.f) ? v : 0.01f * v;
        else if (EPI == 1) v = tanhf(v);
        else if (EPI == 2) v = fmaxf(v, 0.f);
        C[(size_t)row * ldc + col0 + oc] = v;
      }
    }
  }
}

// Column sum / sum-of-squares for BatchNorm statistics.
__global__ __launch_bounds__(256) void col_stats_kernel(
    const float* __restrict__ X, int M, int C, int rowsPerBlock,
    float* __restrict__ stats /* [2*C]: sum, sumsq */)
{
  const int c = blockIdx.x * blockDim.x + threadIdx.x;
  if (c >= C) return;
  const int r0 = blockIdx.y * rowsPerBlock;
  const int r1 = min(r0 + rowsPerBlock, M);
  float s = 0.f, q = 0.f;
  for (int r = r0; r < r1; ++r) {
    const float v = X[(size_t)r * C + c];
    s += v; q += v * v;
  }
  atomicAdd(&stats[c], s);
  atomicAdd(&stats[C + c], q);
}

__global__ void bn_coef_kernel(const float* __restrict__ stats,
                               const float* __restrict__ gamma,
                               const float* __restrict__ beta,
                               int C, float invM,
                               float* __restrict__ coef /* [2*C]: scale, shift */)
{
  const int c = blockIdx.x * blockDim.x + threadIdx.x;
  if (c >= C) return;
  const float mean = stats[c] * invM;
  const float var  = stats[C + c] * invM - mean * mean;
  const float sc   = gamma[c] * rsqrtf(var + 1e-5f);
  coef[c]     = sc;
  coef[C + c] = beta[c] - mean * sc;
}

__global__ void fill_kernel(float* __restrict__ p, float v, size_t n) {
  const size_t i = (size_t)blockIdx.x * blockDim.x + threadIdx.x;
  if (i < n) p[i] = v;
}

// Gather tanh-output, apply BN2 affine per element, atomic-min into agg.
__global__ __launch_bounds__(256) void seg_min_kernel(
    const float* __restrict__ t, const float* __restrict__ coef /* [2*256] */,
    const long long* __restrict__ ei, int E, float* __restrict__ agg)
{
  const long long tid = (long long)blockIdx.x * blockDim.x + threadIdx.x;
  const int e = (int)(tid >> 6);
  if (e >= E) return;
  const int c = (int)(tid & 63) * 4;
  const int s = (int)ei[e];
  const int d = (int)ei[(size_t)E + e];
  const float4 v  = *(const float4*)&t[(size_t)s * 256 + c];
  const float4 sc = *(const float4*)&coef[c];
  const float4 sh = *(const float4*)&coef[256 + c];
  float vals[4] = { v.x*sc.x + sh.x, v.y*sc.y + sh.y,
                    v.z*sc.z + sh.z, v.w*sc.w + sh.w };
  float* base = &agg[(size_t)d * 256 + c];
  #pragma unroll
  for (int j = 0; j < 4; ++j)
    (void)__hip_atomic_fetch_min(&base[j], vals[j],
                                 __ATOMIC_RELAXED, __HIP_MEMORY_SCOPE_AGENT);
}

// Transpose-convert weights: W [K, Nc] f32 -> Wt [Nc, K] bf16.
__global__ void conv_wt_kernel(const float* __restrict__ W,
                               __bf16* __restrict__ Wt, int K, int Nc) {
  const int idx = blockIdx.x * blockDim.x + threadIdx.x;
  if (idx >= K * Nc) return;
  const int k = idx / Nc, n = idx % Nc;
  Wt[(size_t)n * K + k] = (__bf16)W[idx];
}

extern "C" void kernel_launch(void* const* d_in, const int* in_sizes, int n_in,
                              void* d_out, int out_size, void* d_ws, size_t ws_size,
                              hipStream_t stream)
{
  (void)n_in; (void)out_size; (void)ws_size;
  const int IN = 1024, STR = 768, H = 256, OUT = 64;

  const float* x     = (const float*)d_in[0];
  const float* xs    = (const float*)d_in[1];
  /* d_in[2] = x_e, unused by the reference */
  const long long* ei = (const long long*)d_in[3];
  const float* W_sem = (const float*)d_in[4];
  const float* b_sem = (const float*)d_in[5];
  const float* W_str = (const float*)d_in[6];
  const float* b_str = (const float*)d_in[7];
  const float* g1    = (const float*)d_in[8];
  const float* be1   = (const float*)d_in[9];
  const float* Wf    = (const float*)d_in[10];
  const float* bfv   = (const float*)d_in[11];
  const float* g2    = (const float*)d_in[12];
  const float* be2   = (const float*)d_in[13];
  const float* Wc1   = (const float*)d_in[14];
  const float* bc1   = (const float*)d_in[15];
  const float* Wc2   = (const float*)d_in[16];
  const float* bc2   = (const float*)d_in[17];
  float* out = (float*)d_out;

  const int N = in_sizes[0] / IN;
  const int E = in_sizes[3] / 2;

  char* ws = (char*)d_ws;
  size_t off = 0;
  auto alloc = [&](size_t bytes) {
    char* p = ws + off; off = (off + bytes + 255) & ~(size_t)255; return p;
  };
  __bf16* WtS  = (__bf16*)alloc((size_t)IN   * H   * 2);
  __bf16* WtT  = (__bf16*)alloc((size_t)STR  * H   * 2);
  __bf16* WtF  = (__bf16*)alloc((size_t)(2*H)* H   * 2);
  __bf16* Wt1  = (__bf16*)alloc((size_t)H    * H   * 2);
  __bf16* Wt2  = (__bf16*)alloc((size_t)H    * OUT * 2);
  float* stats1 = (float*)alloc((size_t)2 * (2*H) * 4);
  float* coef1  = (float*)alloc((size_t)2 * (2*H) * 4);
  float* stats2 = (float*)alloc((size_t)2 * H * 4);
  float* coef2  = (float*)alloc((size_t)2 * H * 4);
  float* h1 = (float*)alloc((size_t)N * (2*H) * 4); // concat; reused as agg
  float* tb = (float*)alloc((size_t)N * H * 4);     // tanh out; reused as z
  float* agg = h1;
  float* z   = tb;

  const dim3 blk(256);

  // Weight transposes to bf16 + stats zeroing.
  conv_wt_kernel<<<dim3((IN*H   + 255)/256), blk, 0, stream>>>(W_sem, WtS, IN,  H);
  conv_wt_kernel<<<dim3((STR*H  + 255)/256), blk, 0, stream>>>(W_str, WtT, STR, H);
  conv_wt_kernel<<<dim3((2*H*H  + 255)/256), blk, 0, stream>>>(Wf,    WtF, 2*H, H);
  conv_wt_kernel<<<dim3((H*H    + 255)/256), blk, 0, stream>>>(Wc1,   Wt1, H,   H);
  conv_wt_kernel<<<dim3((H*OUT  + 255)/256), blk, 0, stream>>>(Wc2,   Wt2, H,   OUT);
  fill_kernel<<<dim3((2*(2*H) + 255)/256), blk, 0, stream>>>(stats1, 0.f, (size_t)2*(2*H));
  fill_kernel<<<dim3((2*H     + 255)/256), blk, 0, stream>>>(stats2, 0.f, (size_t)2*H);

  const dim3 gmain((unsigned)((N + 63)/64), 2);    // FRAGS=4: 128-col tiles
  const dim3 gout ((unsigned)((N + 63)/64), 1);    // FRAGS=2: 64-col tile

  // Refiners: leaky(x@W+b) into h1 columns [0,256) and [256,512).
  gemm_bf16_wmma<0,0,4><<<gmain, blk, 0, stream>>>(x,  N, IN,  WtS, b_sem, nullptr, nullptr, h1, 2*H, 0);
  gemm_bf16_wmma<0,0,4><<<gmain, blk, 0, stream>>>(xs, N, STR, WtT, b_str, nullptr, nullptr, h1, 2*H, H);

  // BN1 stats -> affine coefficients.
  col_stats_kernel<<<dim3(2, (unsigned)((N + 1023)/1024)), blk, 0, stream>>>(h1, N, 2*H, 1024, stats1);
  bn_coef_kernel<<<dim3(2), blk, 0, stream>>>(stats1, g1, be1, 2*H, 1.f/(float)N, coef1);

  // Fusion: tanh(bn1(h1) @ Wf + bf) -> tb.  (BN folded into A-load.)
  gemm_bf16_wmma<1,1,4><<<gmain, blk, 0, stream>>>(h1, N, 2*H, WtF, bfv, coef1, coef1 + 2*H, tb, H, 0);

  // agg := +inf (h1 region is dead now); BN2 stats over tb.
  fill_kernel<<<dim3((unsigned)(((size_t)N*H + 255)/256)), blk, 0, stream>>>(agg, __builtin_huge_valf(), (size_t)N*H);
  col_stats_kernel<<<dim3(1, (unsigned)((N + 1023)/1024)), blk, 0, stream>>>(tb, N, H, 1024, stats2);
  bn_coef_kernel<<<dim3(1), blk, 0, stream>>>(stats2, g2, be2, H, 1.f/(float)N, coef2);

  // Segment-min of bn2(tb)[src] by dst (affine applied pre-min == min of bn2).
  seg_min_kernel<<<dim3((unsigned)(((size_t)E*64 + 255)/256)), blk, 0, stream>>>(tb, coef2, ei, E, agg);

  // Classifier: relu((inf->0)(agg) @ Wc1 + bc1) @ Wc2 + bc2 -> out.
  gemm_bf16_wmma<2,2,4><<<gmain, blk, 0, stream>>>(agg, N, H, Wt1, bc1, nullptr, nullptr, z, H, 0);
  gemm_bf16_wmma<0,3,2><<<gout,  blk, 0, stream>>>(z,   N, H, Wt2, bc2, nullptr, nullptr, out, OUT, 0);
}
